// Cross_Attention_25383256719681
// MI455X (gfx1250) — matrix operand order repaired and checked by reading, not compiled, not run on hardware
//
#include <hip/hip_runtime.h>
#include <hip/hip_bf16.h>

// ---------------------------------------------------------------------------
// Cross-attention for MI455X (gfx1250).
//  Pre-pass: x,memory -> bf16; Wq/Wk/Wv/Wo -> bf16 transposed [N][K].
//  1) Q = x @ Wq        (bf16 ws [B,H,LT,HD])
//  2) K = memory @ Wk   (fp32 -> d_out k, bf16 ws [B,H,LS,HD])
//  3) V = memory @ Wv   (fp32 -> d_out v, bf16 ws TRANSPOSED [B,H,HD,LS])
//  4) flash attention   (async-to-LDS double-buffered K / V^T / pos tiles)
//  5) score_proj = score_cat @ Wo (fp32 -> d_out)
// All bulk staging uses global_load_async_to_lds_b128 (ASYNCcnt pipeline).
// ---------------------------------------------------------------------------

typedef __attribute__((ext_vector_type(16))) __bf16 v16bf;
typedef __attribute__((ext_vector_type(8)))  float  v8f;
typedef unsigned long long ull;

union Frag {
    v16bf v;
    unsigned u[8];
};

__device__ __forceinline__ unsigned short f2bf(float f) {
    unsigned u = __builtin_bit_cast(unsigned, f);
    u += 0x7FFFu + ((u >> 16) & 1u);   // round-to-nearest-even
    return (unsigned short)(u >> 16);
}

__device__ __forceinline__ void async_b128(unsigned lds_off, const void* gaddr) {
    asm volatile("global_load_async_to_lds_b128 %0, %1, off"
                 :: "v"(lds_off), "v"((ull)(size_t)gaddr) : "memory");
}
__device__ __forceinline__ void wait_async0() {
    asm volatile("s_wait_asynccnt 0" ::: "memory");
}

static constexpr int Bc  = 2;
static constexpr int LTc = 2048;
static constexpr int LSc = 2048;
static constexpr int Dc  = 1024;
static constexpr int Hc  = 16;
static constexpr int HDc = 64;

#define KSTEP 32
#define APAD  40   // 64x32 bf16 tile rows padded to 80B (16B-aligned, bank-spread)
#define KPAD  72   // 64x64 bf16 tile rows padded to 144B
#define PPAD  68   // 64x64 f32 pos tile rows padded to 272B (16B-aligned)

// ---------------------------------------------------------------------------
// Elementwise fp32 -> bf16
// ---------------------------------------------------------------------------
__global__ __launch_bounds__(256)
void cvt_kernel(const float* __restrict__ src, unsigned short* __restrict__ dst,
                int n) {
    int i0 = (blockIdx.x * 256 + threadIdx.x) * 8;
    #pragma unroll
    for (int j = 0; j < 8; ++j) {
        int i = i0 + j;
        if (i < n) dst[i] = f2bf(src[i]);
    }
}

// Transpose-convert weights: WT[n*D+k] = bf16(W[k*D+n])
__global__ __launch_bounds__(256)
void cvtT_kernel(const float* __restrict__ W, unsigned short* __restrict__ WT) {
    int idx = blockIdx.x * 256 + threadIdx.x;        // over D*D, write-coalesced
    int n = idx >> 10, k = idx & (Dc - 1);
    WT[(size_t)n * Dc + k] = f2bf(W[(size_t)k * Dc + n]);
}

// ---------------------------------------------------------------------------
// bf16 GEMM: C[M,N] = A_bf16[M,K] @ BT_bf16[N,K]^T, K = N = 1024, tile 64x64.
// 128 threads = 4 waves; async double-buffered LDS staging.
// mode 0: Q (bf16 head-split)        mode 1: K (bf16 head-split + f32)
// mode 2: V (bf16 TRANSPOSED + f32)  mode 3: O (f32 row-major only)
// ---------------------------------------------------------------------------
__global__ __launch_bounds__(128)
void gemm_kernel(const unsigned short* __restrict__ A,
                 const unsigned short* __restrict__ BT,
                 unsigned short* __restrict__ out_bf,
                 float* __restrict__ out_f32, int L, int mode)
{
    __shared__ __align__(16) unsigned short As[2][64][APAD];
    __shared__ __align__(16) unsigned short Bs[2][64][APAD];

    const int b  = blockIdx.z;
    const int m0 = blockIdx.y * 64;
    const int n0 = blockIdx.x * 64;
    const int tid  = threadIdx.x;
    const int wave = tid >> 5;
    const int lane = tid & 31;
    const int half = lane >> 4;
    const int l16  = lane & 15;

    const unsigned short* Ab = A + (size_t)b * L * Dc;

    v8f acc[4] = {};

    auto issue = [&](int buf, int k0) {
        unsigned abase = (unsigned)(size_t)&As[buf][0][0];
        unsigned bbase = (unsigned)(size_t)&Bs[buf][0][0];
        #pragma unroll
        for (int it = 0; it < 2; ++it) {
            int c = tid + it * 128;                  // 256 chunks of 16B per tile
            int r = c >> 2, o = (c & 3) * 16;
            async_b128(abase + r * (APAD * 2) + o,
                       Ab + (size_t)(m0 + r) * Dc + k0 + (c & 3) * 8);
            async_b128(bbase + r * (APAD * 2) + o,
                       BT + (size_t)(n0 + r) * Dc + k0 + (c & 3) * 8);
        }
    };

    issue(0, 0);
    for (int k0 = 0; k0 < Dc; k0 += KSTEP) {
        const int buf = (k0 >> 5) & 1;
        wait_async0();
        __syncthreads();
        if (k0 + KSTEP < Dc) issue(buf ^ 1, k0 + KSTEP);

        Frag a;
        const int row = wave * 16 + l16;
        #pragma unroll
        for (int v = 0; v < 8; ++v) {
            int kb = ((v & 3) << 1) + ((v >> 2) << 4) + half * 8;
            a.u[v] = *(const unsigned*)&As[buf][row][kb];
        }
        #pragma unroll
        for (int nt = 0; nt < 4; ++nt) {
            Frag bfrag;
            const int col = nt * 16 + l16;
            #pragma unroll
            for (int v = 0; v < 8; ++v) {
                int kb = ((v & 3) << 1) + ((v >> 2) << 4) + half * 8;
                bfrag.u[v] = *(const unsigned*)&Bs[buf][col][kb];
            }
            acc[nt] = __builtin_amdgcn_wmma_f32_16x16x32_bf16(
                false, a.v, false, bfrag.v, (short)0, acc[nt], false, false);
        }
    }

    // epilogue
    #pragma unroll
    for (int nt = 0; nt < 4; ++nt) {
        const int col = n0 + nt * 16 + l16;
        #pragma unroll
        for (int v = 0; v < 8; ++v) {
            const int row = m0 + wave * 16 + v + 8 * half;
            const float val = acc[nt][v];
            if (mode == 3) {
                out_f32[((size_t)b * L + row) * Dc + col] = val;
            } else {
                const int h = col >> 6, hd = col & 63;
                const size_t idx = (((size_t)b * Hc + h) * (size_t)L + row) * HDc + hd;
                if (mode == 2)
                    out_bf[(((size_t)b * Hc + h) * HDc + hd) * (size_t)L + row] = f2bf(val);
                else
                    out_bf[idx] = f2bf(val);
                if (mode >= 1) out_f32[idx] = val;
            }
        }
    }
}

// ---------------------------------------------------------------------------
// Flash attention: block = (b, h, 64 q-rows), 4 waves. K, V^T and pos tiles
// async-copied to LDS, double-buffered; Q fragments direct from global;
// mask stays as direct global loads (L2-resident, reused 32x).
// ---------------------------------------------------------------------------
__global__ __launch_bounds__(128)
void attn_kernel(const unsigned short* __restrict__ Qbf,
                 const unsigned short* __restrict__ Kbf,
                 const unsigned short* __restrict__ VbfT,
                 const float* __restrict__ pos,
                 const float* __restrict__ mask,
                 unsigned short* __restrict__ Sc)
{
    __shared__ __align__(16) unsigned short Ks[2][64][KPAD];  // [s][d]
    __shared__ __align__(16) unsigned short Vs[2][64][KPAD];  // [d][s]
    __shared__ __align__(16) float          PosS[2][64][PPAD];// [q][s] f32
    __shared__ unsigned short Ps[4][16][KPAD];                // per-wave P

    const int qt = blockIdx.x, h = blockIdx.y, b = blockIdx.z;
    const int q0   = qt * 64;
    const int tid  = threadIdx.x;
    const int wave = tid >> 5;
    const int lane = tid & 31;
    const int half = lane >> 4;
    const int l16  = lane & 15;

    const unsigned short* Qg =
        Qbf + (((size_t)b * Hc + h) * LTc + q0 + wave * 16 + l16) * HDc;
    const unsigned short* Kg = Kbf  + (((size_t)b * Hc + h) * LSc) * HDc;
    const unsigned short* Vt = VbfT + (((size_t)b * Hc + h) * HDc) * LSc;
    const float*          Pg = pos  + ((size_t)h * LTc + q0) * LSc;

    Frag qf[2];
    #pragma unroll
    for (int kk = 0; kk < 2; ++kk)
        #pragma unroll
        for (int v = 0; v < 8; ++v) {
            int kb = kk * 32 + ((v & 3) << 1) + ((v >> 2) << 4) + half * 8;
            qf[kk].u[v] = *(const unsigned*)&Qg[kb];
        }

    v8f acc_o[4] = {};
    float mrun[8], lrun[8];
    #pragma unroll
    for (int v = 0; v < 8; ++v) { mrun[v] = -1e30f; lrun[v] = 0.0f; }
    const float scale = 0.125f;  // 1/sqrt(64)

    auto issue = [&](int buf, int s0) {
        unsigned kbase = (unsigned)(size_t)&Ks[buf][0][0];
        unsigned vbase = (unsigned)(size_t)&Vs[buf][0][0];
        unsigned pbase = (unsigned)(size_t)&PosS[buf][0][0];
        #pragma unroll
        for (int it = 0; it < 4; ++it) {
            int c = tid + it * 128;                  // 512 chunks of 16B per tile
            int r = c >> 3, o = (c & 7) * 16;
            async_b128(kbase + r * (KPAD * 2) + o,
                       Kg + (size_t)(s0 + r) * HDc + (c & 7) * 8);
            async_b128(vbase + r * (KPAD * 2) + o,
                       Vt + (size_t)r * LSc + s0 + (c & 7) * 8);
        }
        #pragma unroll
        for (int it = 0; it < 8; ++it) {
            int c = tid + it * 128;                  // 1024 chunks of 16B (f32 tile)
            int r = c >> 4, o = (c & 15) * 16;
            async_b128(pbase + r * (PPAD * 4) + o,
                       Pg + (size_t)r * LSc + s0 + (c & 15) * 4);
        }
    };

    issue(0, 0);
    for (int s0 = 0; s0 < LSc; s0 += 64) {
        const int buf = (s0 >> 6) & 1;
        wait_async0();
        __syncthreads();
        if (s0 + 64 < LSc) issue(buf ^ 1, s0 + 64);

        // S = Q @ K^T
        v8f s_acc[4] = {};
        #pragma unroll
        for (int nt = 0; nt < 4; ++nt) {
            const int scl = nt * 16 + l16;
            #pragma unroll
            for (int kk = 0; kk < 2; ++kk) {
                Frag bk;
                #pragma unroll
                for (int v = 0; v < 8; ++v) {
                    int kb = kk * 32 + ((v & 3) << 1) + ((v >> 2) << 4) + half * 8;
                    bk.u[v] = *(const unsigned*)&Ks[buf][scl][kb];
                }
                s_acc[nt] = __builtin_amdgcn_wmma_f32_16x16x32_bf16(
                    false, qf[kk].v, false, bk.v, (short)0, s_acc[nt], false, false);
            }
        }

        // scale + position bias (LDS) + additive mask (global, L2-resident)
        #pragma unroll
        for (int nt = 0; nt < 4; ++nt) {
            const int scol = s0 + nt * 16 + l16;
            #pragma unroll
            for (int v = 0; v < 8; ++v) {
                const int ql   = wave * 16 + v + 8 * half;   // q-local 0..63
                const int qrow = q0 + ql;
                float bias = PosS[buf][ql][nt * 16 + l16] +
                             mask[(size_t)qrow * LSc + scol];
                s_acc[nt][v] = s_acc[nt][v] * scale + bias;
            }
        }

        // online softmax (per-row stats; 16-lane N-group reduction)
        #pragma unroll
        for (int v = 0; v < 8; ++v) {
            float t = fmaxf(fmaxf(s_acc[0][v], s_acc[1][v]),
                            fmaxf(s_acc[2][v], s_acc[3][v]));
            #pragma unroll
            for (int m = 1; m < 16; m <<= 1) t = fmaxf(t, __shfl_xor(t, m, 32));
            const float newm  = fmaxf(mrun[v], t);
            const float alpha = __expf(mrun[v] - newm);
            mrun[v] = newm;

            float rs = 0.0f;
            #pragma unroll
            for (int nt = 0; nt < 4; ++nt) {
                float p = __expf(s_acc[nt][v] - newm);
                rs += p;
                Ps[wave][v + 8 * half][nt * 16 + l16] = f2bf(p);
            }
            #pragma unroll
            for (int m = 1; m < 16; m <<= 1) rs += __shfl_xor(rs, m, 32);
            lrun[v] = lrun[v] * alpha + rs;
            #pragma unroll
            for (int nt = 0; nt < 4; ++nt) acc_o[nt][v] *= alpha;
        }

        // O += P @ V
        Frag pf[2];
        #pragma unroll
        for (int kk = 0; kk < 2; ++kk)
            #pragma unroll
            for (int v = 0; v < 8; ++v) {
                int kb = kk * 32 + ((v & 3) << 1) + ((v >> 2) << 4) + half * 8;
                pf[kk].u[v] = *(const unsigned*)&Ps[wave][l16][kb];
            }
        #pragma unroll
        for (int nt = 0; nt < 4; ++nt) {
            const int dcol = nt * 16 + l16;
            #pragma unroll
            for (int kk = 0; kk < 2; ++kk) {
                Frag bv;
                #pragma unroll
                for (int v = 0; v < 8; ++v) {
                    int kb = kk * 32 + ((v & 3) << 1) + ((v >> 2) << 4) + half * 8;
                    bv.u[v] = *(const unsigned*)&Vs[buf][dcol][kb];
                }
                acc_o[nt] = __builtin_amdgcn_wmma_f32_16x16x32_bf16(
                    false, pf[kk].v, false, bv.v, (short)0, acc_o[nt], false, false);
            }
        }
    }

    // normalize + store bf16 score_cat [B,LT,D]
    #pragma unroll
    for (int v = 0; v < 8; ++v) {
        const float inv = 1.0f / lrun[v];
        const int qrow = q0 + wave * 16 + v + 8 * half;
        #pragma unroll
        for (int nt = 0; nt < 4; ++nt) {
            const int dcol = nt * 16 + l16;
            Sc[((size_t)b * LTc + qrow) * Dc + h * HDc + dcol] =
                f2bf(acc_o[nt][v] * inv);
        }
    }
}

// ---------------------------------------------------------------------------
extern "C" void kernel_launch(void* const* d_in, const int* in_sizes, int n_in,
                              void* d_out, int out_size, void* d_ws, size_t ws_size,
                              hipStream_t stream) {
    (void)in_sizes; (void)n_in; (void)out_size; (void)ws_size;

    const float* x    = (const float*)d_in[0];
    const float* mem  = (const float*)d_in[1];
    const float* pos  = (const float*)d_in[2];
    const float* mask = (const float*)d_in[3];
    const float* Wq   = (const float*)d_in[4];
    const float* Wk   = (const float*)d_in[5];
    const float* Wv   = (const float*)d_in[6];
    const float* Wo   = (const float*)d_in[7];

    float* out       = (float*)d_out;
    const size_t nSP = (size_t)Bc * LTc * Dc;
    const size_t nKV = (size_t)Bc * Hc * LSc * HDc;
    float* out_k = out + nSP;
    float* out_v = out + nSP + nKV;

    const size_t nBL = (size_t)Bc * LTc * Dc;   // 4M elems
    const size_t nW  = (size_t)Dc * Dc;         // 1M elems
    unsigned short* xbf  = (unsigned short*)d_ws;
    unsigned short* mbf  = xbf  + nBL;
    unsigned short* WqT  = mbf  + nBL;
    unsigned short* WkT  = WqT  + nW;
    unsigned short* WvT  = WkT  + nW;
    unsigned short* WoT  = WvT  + nW;
    unsigned short* Qbf  = WoT  + nW;
    unsigned short* Kbf  = Qbf  + nBL;
    unsigned short* VbfT = Kbf  + nBL;
    unsigned short* Sc   = VbfT + nBL;

    dim3 blk(128);
    dim3 gp(Dc / 64, LTc / 64, Bc);
    dim3 ga(LTc / 64, Hc, Bc);

    // pre-pass: conversions / weight transposes
    cvt_kernel <<<dim3((unsigned)(nBL / (256 * 8))), dim3(256), 0, stream>>>(x,   xbf, (int)nBL);
    cvt_kernel <<<dim3((unsigned)(nBL / (256 * 8))), dim3(256), 0, stream>>>(mem, mbf, (int)nBL);
    cvtT_kernel<<<dim3((unsigned)(nW / 256)), dim3(256), 0, stream>>>(Wq, WqT);
    cvtT_kernel<<<dim3((unsigned)(nW / 256)), dim3(256), 0, stream>>>(Wk, WkT);
    cvtT_kernel<<<dim3((unsigned)(nW / 256)), dim3(256), 0, stream>>>(Wv, WvT);
    cvtT_kernel<<<dim3((unsigned)(nW / 256)), dim3(256), 0, stream>>>(Wo, WoT);

    // projections
    gemm_kernel<<<gp, blk, 0, stream>>>(xbf, WqT, Qbf,  nullptr, LTc, 0);
    gemm_kernel<<<gp, blk, 0, stream>>>(mbf, WkT, Kbf,  out_k,   LSc, 1);
    gemm_kernel<<<gp, blk, 0, stream>>>(mbf, WvT, VbfT, out_v,   LSc, 2);

    // attention
    attn_kernel<<<ga, blk, 0, stream>>>(Qbf, Kbf, VbfT, pos, mask, Sc);

    // output projection
    gemm_kernel<<<gp, blk, 0, stream>>>(Sc, WoT, nullptr, out, LTc, 3);
}
